// StateSpaceTransformer_64922725646922
// MI455X (gfx1250) — compile-verified, matrix operand order checked by reading, not run on hardware
//
#include <hip/hip_runtime.h>
#include <hip/hip_bf16.h>
#include <math.h>

// ---------------------------------------------------------------------------
// Model dims (compile-time constants from the reference)
// ---------------------------------------------------------------------------
#define BB    2
#define NFQ   16
#define PP    64
#define DD    512
#define SS    256
#define HH    8
#define DHH   64
#define LL    4
#define MLPD  2048
#define NN    (NFQ * PP)       // 1024 tokens per stream
#define TN    (2 * NN)         // 2048 ctx tokens
#define INNERD (HH * DHH)      // 512
#define C3    (3 * INNERD)     // 1536

typedef __attribute__((ext_vector_type(16))) _Float16 v16h;
typedef __attribute__((ext_vector_type(8)))  float    v8f;

__device__ __forceinline__ v8f wmma_f16(v16h a, v16h b, v8f c) {
  // emits v_wmma_f32_16x16x32_f16
  return __builtin_amdgcn_wmma_f32_16x16x32_f16(false, a, false, b, (short)0, c,
                                                false, false);
}

// Build a 16-bit A/B operand (16x32 tile) from a row-contiguous f16 chunk.
// Lane layout (wave32): lane holds row/col lane&15; koff=(lane>>4)*8;
// elements 0..7 = K koff..koff+7, elements 8..15 = K 16+koff..16+koff+7.
// Each 8-run is 16 contiguous bytes -> one global_load_b128.
__device__ __forceinline__ v16h load16h(const _Float16* __restrict__ p, int koff) {
  v16h r;
#pragma unroll
  for (int e = 0; e < 8; ++e) {
    r[e]     = p[koff + e];
    r[e + 8] = p[16 + koff + e];
  }
  return r;
}

// ---------------------------------------------------------------------------
// f32 -> f16 cast (4 elems / thread), for one-time weight conversion
// ---------------------------------------------------------------------------
__global__ void cast4_kernel(const float* __restrict__ in,
                             _Float16* __restrict__ out, int n4) {
  const int i = blockIdx.x * blockDim.x + threadIdx.x;
  if (i >= n4) return;
  const float4 v = ((const float4*)in)[i];
  out[4 * i + 0] = (_Float16)v.x;
  out[4 * i + 1] = (_Float16)v.y;
  out[4 * i + 2] = (_Float16)v.z;
  out[4 * i + 3] = (_Float16)v.w;
}

// ---------------------------------------------------------------------------
// LayerNorm over D=512 (one block / row, 128 threads); f32 and/or f16 output
// ---------------------------------------------------------------------------
__global__ void ln_kernel(const float* __restrict__ in,
                          const float* __restrict__ g,
                          const float* __restrict__ b,
                          float* outf, _Float16* outh) {
  const int row = blockIdx.x;
  const float* x = in + (size_t)row * DD;
  float s = 0.f, s2 = 0.f;
  for (int i = threadIdx.x; i < DD; i += blockDim.x) {
    float v = x[i]; s += v; s2 += v * v;
  }
  __shared__ float r1[128], r2[128];
  r1[threadIdx.x] = s; r2[threadIdx.x] = s2;
  __syncthreads();
  for (int o = 64; o > 0; o >>= 1) {
    if ((int)threadIdx.x < o) { r1[threadIdx.x] += r1[threadIdx.x + o];
                                r2[threadIdx.x] += r2[threadIdx.x + o]; }
    __syncthreads();
  }
  const float mean = r1[0] * (1.0f / DD);
  const float var  = r2[0] * (1.0f / DD) - mean * mean;
  const float inv  = rsqrtf(var + 1e-5f);
  for (int i = threadIdx.x; i < DD; i += blockDim.x) {
    const float v = (x[i] - mean) * inv * g[i] + b[i];
    const size_t idx = (size_t)row * DD + i;
    if (outf) outf[idx] = v;
    if (outh) outh[idx] = (_Float16)v;
  }
}

// fused concat([M, xn], axis=1) + LayerNorm -> ctx [B, 2N, D] (f32)
__global__ void fuse_ln_kernel(const float* __restrict__ M,
                               const float* __restrict__ xn,
                               const float* __restrict__ g,
                               const float* __restrict__ b,
                               float* __restrict__ ctx) {
  const int row = blockIdx.x;          // [0, B*TN)
  const int bb = row / TN;
  const int i  = row % TN;
  const float* x = (i < NN) ? (M  + ((size_t)bb * NN + i) * DD)
                            : (xn + ((size_t)bb * NN + (i - NN)) * DD);
  float* y = ctx + (size_t)row * DD;
  float s = 0.f, s2 = 0.f;
  for (int k = threadIdx.x; k < DD; k += blockDim.x) {
    float v = x[k]; s += v; s2 += v * v;
  }
  __shared__ float r1[128], r2[128];
  r1[threadIdx.x] = s; r2[threadIdx.x] = s2;
  __syncthreads();
  for (int o = 64; o > 0; o >>= 1) {
    if ((int)threadIdx.x < o) { r1[threadIdx.x] += r1[threadIdx.x + o];
                                r2[threadIdx.x] += r2[threadIdx.x + o]; }
    __syncthreads();
  }
  const float mean = r1[0] * (1.0f / DD);
  const float var  = r2[0] * (1.0f / DD) - mean * mean;
  const float inv  = rsqrtf(var + 1e-5f);
  for (int k = threadIdx.x; k < DD; k += blockDim.x)
    y[k] = (x[k] - mean) * inv * g[k] + b[k];
}

// ---------------------------------------------------------------------------
// SSM diagonal scan over NF frames. One thread per (b,p,s). f16 in/out,
// f32 recurrence.
// ---------------------------------------------------------------------------
__global__ void ssm_scan_kernel(const _Float16* __restrict__ Ux,
                                const float* __restrict__ log_tau,
                                _Float16* __restrict__ Hs) {
  const int idx = blockIdx.x * blockDim.x + threadIdx.x;
  if (idx >= BB * PP * SS) return;
  const int s = idx & (SS - 1);
  const int p = (idx >> 8) & (PP - 1);
  const int b = idx >> 14;
  const float lt  = log_tau[s];
  float tau = (lt > 20.f) ? lt : log1pf(__expf(lt));   // softplus
  tau += 1e-4f;
  const float A  = __expf(-1.0f / tau);                // DT = 1.0
  const float Bb = 1.0f - A;
  float h = 0.f;
  const size_t base = ((size_t)b * NN + p) * SS + s;   // row = b*N + f*P + p
  const size_t fstride = (size_t)PP * SS;
#pragma unroll
  for (int f = 0; f < NFQ; ++f) {
    const size_t off = base + (size_t)f * fstride;
    h = A * h + Bb * (float)Ux[off];
    Hs[off] = (_Float16)h;
  }
}

// ---------------------------------------------------------------------------
// Generic WMMA GEMM: acc = A[M,K] @ W[N,K]^T  (f16 inputs, f32 accum)
// epilogue: +bias, optional exact GELU, then write f32 (+residual) and/or f16.
// One wave computes a 16x64 tile. K must be a multiple of 32.
// ---------------------------------------------------------------------------
__global__ void __launch_bounds__(128)
wmma_gemm_kernel(const _Float16* __restrict__ A, const _Float16* __restrict__ W,
                 const float* __restrict__ bias, const float* resid,
                 float* Cf, _Float16* Ch,
                 int Mdim, int Ndim, int Kdim, int gelu) {
  const int wave  = (blockIdx.x * blockDim.x + threadIdx.x) >> 5;
  const int lane  = threadIdx.x & 31;
  const int tilesM = Mdim >> 4, tilesN = Ndim >> 6;
  if (wave >= tilesM * tilesN) return;
  const int tm = wave % tilesM, tn = wave / tilesM;
  const int m0 = tm << 4, n0 = tn << 6;
  const int halfrow = lane & 15;
  const int koff    = (lane >> 4) << 3;

  const _Float16* Arow = A + (size_t)(m0 + halfrow) * Kdim;
  const _Float16* W0 = W + (size_t)(n0 +  0 + halfrow) * Kdim;
  const _Float16* W1 = W + (size_t)(n0 + 16 + halfrow) * Kdim;
  const _Float16* W2 = W + (size_t)(n0 + 32 + halfrow) * Kdim;
  const _Float16* W3 = W + (size_t)(n0 + 48 + halfrow) * Kdim;

  v8f acc0 = {}, acc1 = {}, acc2 = {}, acc3 = {};
  for (int k0 = 0; k0 < Kdim; k0 += 32) {
    __builtin_prefetch(Arow + k0 + 128, 0, 1);   // global_prefetch_b8
    const v16h a = load16h(Arow + k0, koff);
    acc0 = wmma_f16(a, load16h(W0 + k0, koff), acc0);
    acc1 = wmma_f16(a, load16h(W1 + k0, koff), acc1);
    acc2 = wmma_f16(a, load16h(W2 + k0, koff), acc2);
    acc3 = wmma_f16(a, load16h(W3 + k0, koff), acc3);
  }

  const int rbase = m0 + ((lane >> 4) << 3);   // C layout: VGPR v -> row v / v+8
  v8f accs[4] = {acc0, acc1, acc2, acc3};
#pragma unroll
  for (int t = 0; t < 4; ++t) {
    const int col = n0 + t * 16 + halfrow;
#pragma unroll
    for (int v = 0; v < 8; ++v) {
      float val = accs[t][v];
      if (bias)  val += bias[col];
      if (gelu)  val = 0.5f * val * (1.0f + erff(val * 0.70710678118f));
      const size_t idx = (size_t)(rbase + v) * Ndim + col;
      if (Cf) {
        float o = val;
        if (resid) o += resid[idx];
        Cf[idx] = o;
      }
      if (Ch) Ch[idx] = (_Float16)val;
    }
  }
}

// ---------------------------------------------------------------------------
// Flash attention, frame-block-sparse mask. One wave per (b, h, 16-query tile).
// qkv: [B, TN, 3*INNER] f16, out: [B, TN, INNER] f16.
// Mask: allowed iff frame(j % N) <= frame(i % N); 32-key blocks are uniform.
// ---------------------------------------------------------------------------
__global__ void __launch_bounds__(32)
attn_kernel(const _Float16* __restrict__ qkv, _Float16* __restrict__ out) {
  const int lane = threadIdx.x;
  const int tid  = blockIdx.x;
  const int qt = tid & 127;              // TN/16 = 128 query tiles
  const int h  = (tid >> 7) & (HH - 1);
  const int b  = tid >> 10;
  const int i0 = qt << 4;
  const int frame_q = (i0 & (NN - 1)) >> 6;     // P = 64
  const int halfrow = lane & 15;
  const int koff    = (lane >> 4) << 3;

  // Q as two A-operands (DH=64 -> K chunks 0..31, 32..63)
  const _Float16* qrow = qkv + ((size_t)(b * TN + i0 + halfrow)) * C3 + h * DHH;
  const v16h aQ0 = load16h(qrow, koff);
  const v16h aQ1 = load16h(qrow + 32, koff);

  float mrow[8], lrow[8];
  v8f O[4] = {v8f{}, v8f{}, v8f{}, v8f{}};
#pragma unroll
  for (int v = 0; v < 8; ++v) { mrow[v] = -1e30f; lrow[v] = 0.f; }

  __shared__ float sP[16][36];   // padded 16x32 prob tile for C->A transpose

  for (int j0 = 0; j0 < TN; j0 += 32) {
    const int frame_k = (j0 & (NN - 1)) >> 6;
    if (frame_k > frame_q) continue;   // fully masked block (uniform branch)

    // S[16x32] = Q . K^T  (keys j0..j0+31)
    const _Float16* kb0 = qkv + (size_t)(b * TN + j0 + halfrow) * C3 + INNERD + h * DHH;
    const _Float16* kb1 = kb0 + (size_t)16 * C3;
    v8f s0 = {}, s1 = {};
    s0 = wmma_f16(aQ0, load16h(kb0,      koff), s0);
    s0 = wmma_f16(aQ1, load16h(kb0 + 32, koff), s0);
    s1 = wmma_f16(aQ0, load16h(kb1,      koff), s1);
    s1 = wmma_f16(aQ1, load16h(kb1 + 32, koff), s1);

    // online softmax: rows live in WMMA C layout (element v = row v / v+8 per half)
    float pscale[8];
#pragma unroll
    for (int v = 0; v < 8; ++v) {
      float x0 = s0[v] * 0.125f;       // DH^-0.5
      float x1 = s1[v] * 0.125f;
      float mx = fmaxf(x0, x1);
#pragma unroll
      for (int msk = 1; msk < 16; msk <<= 1)
        mx = fmaxf(mx, __shfl_xor(mx, msk, 32));  // reduce within 16-lane half
      const float mnew = fmaxf(mrow[v], mx);
      const float sc = __expf(mrow[v] - mnew);
      mrow[v] = mnew;
      const float p0 = __expf(x0 - mnew);
      const float p1 = __expf(x1 - mnew);
      float rs = p0 + p1;
#pragma unroll
      for (int msk = 1; msk < 16; msk <<= 1)
        rs += __shfl_xor(rs, msk, 32);
      lrow[v] = lrow[v] * sc + rs;
      s0[v] = p0; s1[v] = p1;
      pscale[v] = sc;
    }
#pragma unroll
    for (int t = 0; t < 4; ++t)
#pragma unroll
      for (int v = 0; v < 8; ++v) O[t][v] *= pscale[v];

    // transpose P (C layout -> A layout) through LDS
    __syncthreads();
#pragma unroll
    for (int v = 0; v < 8; ++v) {
      const int r = v + ((lane >> 4) << 3);
      sP[r][halfrow]      = s0[v];
      sP[r][16 + halfrow] = s1[v];
    }
    __syncthreads();
    v16h aP;
#pragma unroll
    for (int e = 0; e < 8; ++e) {
      aP[e]     = (_Float16)sP[halfrow][koff + e];
      aP[e + 8] = (_Float16)sP[halfrow][16 + koff + e];
    }

    // O[16x64] += P[16x32] . V[32x64]
    const _Float16* vbase = qkv + (size_t)(b * TN + j0) * C3 + 2 * INNERD + h * DHH;
#pragma unroll
    for (int t = 0; t < 4; ++t) {
      const int d = t * 16 + halfrow;
      v16h bV;
#pragma unroll
      for (int e = 0; e < 8; ++e) {
        bV[e]     = vbase[(size_t)(koff + e) * C3 + d];
        bV[e + 8] = vbase[(size_t)(16 + koff + e) * C3 + d];
      }
      O[t] = wmma_f16(aP, bV, O[t]);
    }
  }

  // normalize + scatter heads back to [B, TN, INNER] (f16)
#pragma unroll
  for (int t = 0; t < 4; ++t) {
    const int d = t * 16 + halfrow;
#pragma unroll
    for (int v = 0; v < 8; ++v) {
      const int row = i0 + v + ((lane >> 4) << 3);
      out[(size_t)(b * TN + row) * INNERD + h * DHH + d] =
          (_Float16)(O[t][v] / lrow[v]);
    }
  }
}

// ---------------------------------------------------------------------------
// final output = ctx[:, N:, :]
// ---------------------------------------------------------------------------
__global__ void copy_out_kernel(const float* __restrict__ ctx,
                                float* __restrict__ out) {
  const int idx = blockIdx.x * blockDim.x + threadIdx.x;
  if (idx >= BB * NN * DD) return;
  const int b   = idx >> 19;           // N*D = 524288 = 2^19
  const int rem = idx & ((NN * DD) - 1);
  out[idx] = ctx[(size_t)b * TN * DD + (size_t)NN * DD + rem];
}

// ---------------------------------------------------------------------------
// host launch
// ---------------------------------------------------------------------------
static inline void launch_cast(const float* in, _Float16* out, int n,
                               hipStream_t stream) {
  const int n4 = n >> 2;
  cast4_kernel<<<(n4 + 255) / 256, 256, 0, stream>>>(in, out, n4);
}

static inline void launch_gemm(const _Float16* A, const _Float16* W,
                               const float* bias, const float* resid,
                               float* Cf, _Float16* Ch,
                               int M, int N, int K, int gelu, hipStream_t stream) {
  const int waves  = (M >> 4) * (N >> 6);
  const int blocks = (waves + 3) / 4;   // 4 waves / 128-thread block
  wmma_gemm_kernel<<<blocks, 128, 0, stream>>>(A, W, bias, resid, Cf, Ch,
                                               M, N, K, gelu);
}

extern "C" void kernel_launch(void* const* d_in, const int* in_sizes, int n_in,
                              void* d_out, int out_size, void* d_ws, size_t ws_size,
                              hipStream_t stream) {
  (void)in_sizes; (void)n_in; (void)out_size; (void)ws_size;

  const float* x         = (const float*)d_in[0];
  const float* ln_in_g   = (const float*)d_in[1];
  const float* ln_in_b   = (const float*)d_in[2];
  const float* U_w       = (const float*)d_in[3];   // [S, D]
  const float* C_w       = (const float*)d_in[4];   // [D, S]
  const float* log_tau   = (const float*)d_in[5];
  const float* ln_fuse_g = (const float*)d_in[6];
  const float* ln_fuse_b = (const float*)d_in[7];
  const float* attn_ln_g = (const float*)d_in[8];   // [L, D]
  const float* attn_ln_b = (const float*)d_in[9];
  const float* qkv_w     = (const float*)d_in[10];  // [L, 3*INNER, D]
  const float* out_w     = (const float*)d_in[11];  // [L, D, INNER]
  const float* out_b     = (const float*)d_in[12];  // [L, D]
  const float* ff_ln_g   = (const float*)d_in[13];
  const float* ff_ln_b   = (const float*)d_in[14];
  const float* w1        = (const float*)d_in[15];  // [L, MLP, D]
  const float* b1        = (const float*)d_in[16];  // [L, MLP]
  const float* w2        = (const float*)d_in[17];  // [L, D, MLP]
  const float* b2        = (const float*)d_in[18];  // [L, D]
  float* out = (float*)d_out;

  // ---- workspace carve (bytes) ----
  char* base = (char*)d_ws;
  // f16 weight mirrors (cast once per launch)
  _Float16* wh_U   = (_Float16*)base; base += (size_t)SS * DD * 2;
  _Float16* wh_C   = (_Float16*)base; base += (size_t)DD * SS * 2;
  _Float16* wh_qkv = (_Float16*)base; base += (size_t)LL * C3 * DD * 2;
  _Float16* wh_out = (_Float16*)base; base += (size_t)LL * DD * INNERD * 2;
  _Float16* wh_w1  = (_Float16*)base; base += (size_t)LL * MLPD * DD * 2;
  _Float16* wh_w2  = (_Float16*)base; base += (size_t)LL * DD * MLPD * 2;
  // activations
  float*    xn_f   = (float*)base;    base += (size_t)BB * NN * DD * 4;
  _Float16* xn_h   = (_Float16*)base; base += (size_t)BB * NN * DD * 2;
  _Float16* ux_h   = (_Float16*)base; base += (size_t)BB * NN * SS * 2;
  _Float16* hs_h   = (_Float16*)base; base += (size_t)BB * NN * SS * 2;
  float*    m_f    = (float*)base;    base += (size_t)BB * NN * DD * 4;
  float*    ctx_f  = (float*)base;    base += (size_t)BB * TN * DD * 4;
  _Float16* tmp_h  = (_Float16*)base; base += (size_t)BB * TN * DD * 2;
  _Float16* qkv_h  = (_Float16*)base; base += (size_t)BB * TN * C3 * 2;
  _Float16* ao_h   = (_Float16*)base; base += (size_t)BB * TN * INNERD * 2;
  _Float16* mlp_h  = (_Float16*)base; base += (size_t)BB * TN * MLPD * 2;

  // 0) one-time weight casts (f32 -> f16)
  launch_cast(U_w,   wh_U,   SS * DD,              stream);
  launch_cast(C_w,   wh_C,   DD * SS,              stream);
  launch_cast(qkv_w, wh_qkv, LL * C3 * DD,         stream);
  launch_cast(out_w, wh_out, LL * DD * INNERD,     stream);
  launch_cast(w1,    wh_w1,  LL * MLPD * DD,       stream);
  launch_cast(w2,    wh_w2,  LL * DD * MLPD,       stream);

  // 1) xn = LN(x)  (f32 for fuse, f16 for GEMM)
  ln_kernel<<<BB * NN, 128, 0, stream>>>(x, ln_in_g, ln_in_b, xn_f, xn_h);

  // 2) Ux = xn @ U_w^T   [2048,512] x [512,256] -> f16
  launch_gemm(xn_h, wh_U, nullptr, nullptr, nullptr, ux_h,
              BB * NN, SS, DD, 0, stream);

  // 3) diagonal scan over frames -> Hs (f16)
  ssm_scan_kernel<<<(BB * PP * SS + 255) / 256, 256, 0, stream>>>(ux_h, log_tau,
                                                                  hs_h);

  // 4) M = Hs @ C_w^T    [2048,256] x [256,512] -> f32
  launch_gemm(hs_h, wh_C, nullptr, nullptr, m_f, nullptr,
              BB * NN, DD, SS, 0, stream);

  // 5) ctx = LN(concat([M, xn], axis=1))
  fuse_ln_kernel<<<BB * TN, 128, 0, stream>>>(m_f, xn_f, ln_fuse_g, ln_fuse_b,
                                              ctx_f);

  // 6) transformer layers
  for (int l = 0; l < LL; ++l) {
    ln_kernel<<<BB * TN, 128, 0, stream>>>(ctx_f, attn_ln_g + l * DD,
                                           attn_ln_b + l * DD, nullptr, tmp_h);
    launch_gemm(tmp_h, wh_qkv + (size_t)l * C3 * DD, nullptr, nullptr,
                nullptr, qkv_h, BB * TN, C3, DD, 0, stream);
    attn_kernel<<<BB * HH * (TN / 16), 32, 0, stream>>>(qkv_h, ao_h);
    launch_gemm(ao_h, wh_out + (size_t)l * DD * INNERD, out_b + (size_t)l * DD,
                ctx_f, ctx_f, nullptr, BB * TN, DD, INNERD, 0, stream);
    ln_kernel<<<BB * TN, 128, 0, stream>>>(ctx_f, ff_ln_g + l * DD,
                                           ff_ln_b + l * DD, nullptr, tmp_h);
    launch_gemm(tmp_h, wh_w1 + (size_t)l * MLPD * DD, b1 + (size_t)l * MLPD,
                nullptr, nullptr, mlp_h, BB * TN, MLPD, DD, 1 /*gelu*/, stream);
    launch_gemm(mlp_h, wh_w2 + (size_t)l * DD * MLPD, b2 + (size_t)l * DD,
                ctx_f, ctx_f, nullptr, BB * TN, DD, MLPD, 0, stream);
  }

  // 7) out = ctx[:, N:, :]
  copy_out_kernel<<<(BB * NN * DD + 255) / 256, 256, 0, stream>>>(ctx_f, out);
}
